// InteractionBlock_58222576664742
// MI455X (gfx1250) — matrix operand order repaired and checked by reading, not compiled, not run on hardware
//
#include <hip/hip_runtime.h>

#define N_NODES   50000
#define N_EDGES   500000
#define MUL0      16
#define MUL1      8
#define DIM       40          // MUL0 + 3*MUL1
#define N_BASIS   10
#define MID       32
#define WNUM      576
#define NTILE     36          // WNUM / 16
#define EPW       16          // edges per wave (M of WMMA)
#define WPB       8           // waves per block

typedef __attribute__((ext_vector_type(16))) __bf16 v16bf;
typedef __attribute__((ext_vector_type(8)))  float  v8f;

__device__ __forceinline__ float silu_f(float t) { return t / (1.0f + __expf(-t)); }

// ---------------------------------------------------------------------------
// Prep: convert W2 [32 x 576] f32 -> bf16 WMMA B-fragments (1/sqrt(32) folded).
// Fragment memory order: idx = (t*32 + lane)*8 + r, packed pair {K=k0, K=k0+1}
// with N = lane&15, k0 = 16*(lane>=16) + 2r  (contiguous k order, same as the A fragment).
// ---------------------------------------------------------------------------
__global__ void prep_w2_frags(const float* __restrict__ W2, unsigned* __restrict__ frag) {
    int idx = blockIdx.x * blockDim.x + threadIdx.x;
    if (idx >= NTILE * 32 * 8) return;
    int r = idx & 7;
    int l = (idx >> 3) & 31;
    int t = idx >> 8;
    int n  = l & 15;
    int k0 = ((l >= 16) ? 16 : 0) + 2 * r;
    int col = t * 16 + n;
    const float sc = 0.17677669529663687f;   // 1/sqrt(MID)
    union { __bf16 b[2]; unsigned u; } p;
    p.b[0] = (__bf16)(W2[(k0 + 0) * WNUM + col] * sc);
    p.b[1] = (__bf16)(W2[(k0 + 1) * WNUM + col] * sc);
    frag[idx] = p.u;
}

__device__ __forceinline__ v16bf load_bfrag(const unsigned* __restrict__ p) {
    union { unsigned u[8]; v16bf v; } b;
#pragma unroll
    for (int r = 0; r < 8; ++r) b.u[r] = p[r];
    return b.v;
}

// ---------------------------------------------------------------------------
// Fused edge kernel: per wave, 16 edges.
//   h = silu(emb@W1/sqrt(10))  -> bf16 A fragment (16x32)
//   w tile (16x16) = WMMA(A, W2frag[t])   for t = 0..35, consumed immediately
//   scatter-add 40 outputs per edge into node accumulator (d_out).
// ---------------------------------------------------------------------------
__global__ void __launch_bounds__(256) edge_kernel(
    const float* __restrict__ x,
    const float* __restrict__ edge_sh,
    const float* __restrict__ emb,
    const float* __restrict__ W1,
    const unsigned* __restrict__ w2frag,
    const int* __restrict__ edge_src,
    const int* __restrict__ edge_dst,
    float* __restrict__ nodeacc)
{
    __shared__ unsigned s_frag[NTILE * 32 * 8];        // 36 KB
    __shared__ float s_w1[N_BASIS * MID];              // 1.25 KB
    __shared__ float s_s  [WPB][16][16];               // gathered scalars
    __shared__ float s_v  [WPB][16][8][3];             // gathered vectors
    __shared__ float s_vd [WPB][16][8];                // vdot = v.sh1 / sqrt(3)
    __shared__ float s_sh0[WPB][16];
    __shared__ float s_sh1[WPB][16][3];

    const int tid = threadIdx.x;
    for (int i = tid; i < NTILE * 32 * 8; i += 256) s_frag[i] = w2frag[i];
    for (int i = tid; i < N_BASIS * MID;  i += 256) s_w1[i]  = W1[i];
    __syncthreads();

    const int wave = tid >> 5;
    const int l    = tid & 31;
    const int n    = l & 15;      // N index of D tile / edge-in-tile for staging
    const int half = l >> 4;

    const int NT   = (N_EDGES + EPW - 1) / EPW;
    const int tile = blockIdx.x * WPB + wave;
    if (tile >= NT) return;                       // wave-uniform exit
    const long base = (long)tile * EPW;

    const int  m  = n;                            // this lane stages edge m
    long e        = base + m;
    bool evalid   = (e < N_EDGES);
    long ec       = evalid ? e : (long)(N_EDGES - 1);

    // ---- MLP layer 1 + silu, build A fragment (16-bit A 16x32 layout) ----
    float er[N_BASIS];
#pragma unroll
    for (int j = 0; j < N_BASIS; ++j) er[j] = emb[ec * N_BASIS + j];
    const float inv_sqrt_nb = 0.31622776601683794f;   // 1/sqrt(10)
    const int kb1 = half ? 16 : 0;
    const int kb2 = half ? 24 : 8;
    v16bf A;
#pragma unroll
    for (int idx = 0; idx < 16; ++idx) {
        int k = (idx < 8) ? (kb1 + idx) : (kb2 + idx - 8);
        float t = 0.f;
#pragma unroll
        for (int j = 0; j < N_BASIS; ++j) t += er[j] * s_w1[j * MID + k];
        A[idx] = (__bf16)silu_f(t * inv_sqrt_nb);
    }

    // ---- stage gathered per-edge data into this wave's LDS region ----
    int src = edge_src[ec];
    if (half == 0) {
#pragma unroll
        for (int u = 0; u < 16; ++u) s_s[wave][m][u] = x[(long)src * DIM + u];
    } else {
        float sh[4];
#pragma unroll
        for (int i = 0; i < 4; ++i) sh[i] = edge_sh[ec * 4 + i];
        s_sh0[wave][m] = sh[0];
#pragma unroll
        for (int i = 0; i < 3; ++i) s_sh1[wave][m][i] = sh[1 + i];
        const float inv_sqrt3 = 0.5773502691896258f;
#pragma unroll
        for (int u = 0; u < 8; ++u) {
            float vd = 0.f;
#pragma unroll
            for (int i = 0; i < 3; ++i) {
                float vv = x[(long)src * DIM + MUL0 + u * 3 + i];
                s_v[wave][m][u][i] = vv;
                vd += vv * sh[1 + i];
            }
            s_vd[wave][m][u] = vd * inv_sqrt3;
        }
    }
    // intra-wave LDS visibility (LDS is in-order within a wave)
    asm volatile("s_wait_dscnt 0" ::: "memory");
    __builtin_amdgcn_wave_barrier();

    // ---- per-lane preloads for the 8 edges this lane owns in D ----
    float sh0r[8]; int dstc[8]; bool vr[8];
#pragma unroll
    for (int r = 0; r < 8; ++r) {
        int mr = r + 8 * half;
        sh0r[r] = s_sh0[wave][mr];
        long ee = base + mr;
        vr[r]   = (ee < N_EDGES);
        dstc[r] = edge_dst[vr[r] ? ee : (long)(N_EDGES - 1)];
    }

    v8f czero = {};
    const unsigned* fbase = &s_frag[(size_t)l * 8];

    // tiles 0..15 : W_ss0, weight = s[u]*sh0 ; tiles 16..23 : W_vv0, weight = vdot
    float acc0[8];
#pragma unroll
    for (int r = 0; r < 8; ++r) acc0[r] = 0.f;
    for (int t = 0; t < 16; ++t) {
        v16bf B = load_bfrag(fbase + (size_t)t * 256);
        v8f d = __builtin_amdgcn_wmma_f32_16x16x32_bf16(false, A, false, B,
                                                        (short)0, czero, false, false);
#pragma unroll
        for (int r = 0; r < 8; ++r) {
            int mr = r + 8 * half;
            acc0[r] += s_s[wave][mr][t] * sh0r[r] * d[r];
        }
    }
    for (int t = 16; t < 24; ++t) {
        v16bf B = load_bfrag(fbase + (size_t)t * 256);
        v8f d = __builtin_amdgcn_wmma_f32_16x16x32_bf16(false, A, false, B,
                                                        (short)0, czero, false, false);
#pragma unroll
        for (int r = 0; r < 8; ++r) {
            int mr = r + 8 * half;
            acc0[r] += s_vd[wave][mr][t - 16] * d[r];
        }
    }

    // tiles 24..31 : W_sv1 (u = 2j + (n>=8), k = n&7), weight = s[u]
    float accsv[8];
#pragma unroll
    for (int r = 0; r < 8; ++r) accsv[r] = 0.f;
    for (int j = 0; j < 8; ++j) {
        v16bf B = load_bfrag(fbase + (size_t)(24 + j) * 256);
        v8f d = __builtin_amdgcn_wmma_f32_16x16x32_bf16(false, A, false, B,
                                                        (short)0, czero, false, false);
        int u = 2 * j + ((n >= 8) ? 1 : 0);
#pragma unroll
        for (int r = 0; r < 8; ++r) {
            int mr = r + 8 * half;
            accsv[r] += s_s[wave][mr][u] * d[r];
        }
    }

    // tiles 32..35 : W_vs1 (u = 2j + (n>=8), k = n&7), weight = v[u][i]
    float accvs[8][3];
#pragma unroll
    for (int r = 0; r < 8; ++r)
#pragma unroll
        for (int i = 0; i < 3; ++i) accvs[r][i] = 0.f;
    for (int j = 0; j < 4; ++j) {
        v16bf B = load_bfrag(fbase + (size_t)(32 + j) * 256);
        v8f d = __builtin_amdgcn_wmma_f32_16x16x32_bf16(false, A, false, B,
                                                        (short)0, czero, false, false);
        int u = 2 * j + ((n >= 8) ? 1 : 0);
#pragma unroll
        for (int r = 0; r < 8; ++r) {
            int mr = r + 8 * half;
#pragma unroll
            for (int i = 0; i < 3; ++i) accvs[r][i] += s_v[wave][mr][u][i] * d[r];
        }
    }

    // ---- epilogue: scale + scatter-add ----
    const float A0c = 0.2041241452319315f;           // sqrt(1/FAN_IN)
    const float C1  = 0.2041241452319315f;           // A1 * 1/sqrt(3) = sqrt(1/24)

#pragma unroll
    for (int r = 0; r < 8; ++r)
        if (vr[r]) atomicAdd(&nodeacc[(long)dstc[r] * DIM + n], A0c * acc0[r]);

    // combine even/odd-u halves (lane n with lane n+8)
    float svt[8], vst[8][3];
#pragma unroll
    for (int r = 0; r < 8; ++r) {
        svt[r] = accsv[r] + __shfl_xor(accsv[r], 8, 32);
#pragma unroll
        for (int i = 0; i < 3; ++i)
            vst[r][i] = accvs[r][i] + __shfl_xor(accvs[r][i], 8, 32);
    }
    if (n < 8) {
#pragma unroll
        for (int r = 0; r < 8; ++r) {
            if (!vr[r]) continue;
            int mr = r + 8 * half;
#pragma unroll
            for (int i = 0; i < 3; ++i) {
                float val = C1 * (svt[r] * s_sh1[wave][mr][i] + vst[r][i] * sh0r[r]);
                atomicAdd(&nodeacc[(long)dstc[r] * DIM + MUL0 + n * 3 + i], val);
            }
        }
    }
}

// ---------------------------------------------------------------------------
// LayerNorm: node /= sqrt(avg_num_neighbors); LN over 40 dims (in place).
// ---------------------------------------------------------------------------
__global__ void ln_kernel(float* __restrict__ node,
                          const float* __restrict__ gamma,
                          const float* __restrict__ beta,
                          const int* __restrict__ avgp)
{
    int i = blockIdx.x * blockDim.x + threadIdx.x;
    if (i >= N_NODES) return;
    float inv = rsqrtf((float)avgp[0]);
    float v[DIM];
    float mu = 0.f;
#pragma unroll
    for (int d = 0; d < DIM; ++d) { v[d] = node[(long)i * DIM + d] * inv; mu += v[d]; }
    mu *= (1.0f / DIM);
    float var = 0.f;
#pragma unroll
    for (int d = 0; d < DIM; ++d) { float c = v[d] - mu; var += c * c; }
    var *= (1.0f / DIM);
    float rs = rsqrtf(var + 1e-5f);
#pragma unroll
    for (int d = 0; d < DIM; ++d)
        node[(long)i * DIM + d] = (v[d] - mu) * rs * gamma[d] + beta[d];
}

extern "C" void kernel_launch(void* const* d_in, const int* in_sizes, int n_in,
                              void* d_out, int out_size, void* d_ws, size_t ws_size,
                              hipStream_t stream) {
    const float* x     = (const float*)d_in[0];
    const float* esh   = (const float*)d_in[1];
    const float* emb   = (const float*)d_in[2];
    const float* W1    = (const float*)d_in[3];
    const float* W2    = (const float*)d_in[4];
    const float* gamma = (const float*)d_in[5];
    const float* beta  = (const float*)d_in[6];
    const int*   esrc  = (const int*)d_in[7];
    const int*   edst  = (const int*)d_in[8];
    const int*   avg   = (const int*)d_in[9];
    float* out = (float*)d_out;
    unsigned* w2frag = (unsigned*)d_ws;            // 36 KB scratch

    hipMemsetAsync(d_out, 0, (size_t)N_NODES * DIM * sizeof(float), stream);

    prep_w2_frags<<<dim3((NTILE * 32 * 8 + 255) / 256), dim3(256), 0, stream>>>(W2, w2frag);

    const int NT = (N_EDGES + EPW - 1) / EPW;
    const int blocks = (NT + WPB - 1) / WPB;
    edge_kernel<<<dim3(blocks), dim3(256), 0, stream>>>(x, esh, emb, W1, w2frag,
                                                        esrc, edst, out);

    ln_kernel<<<dim3((N_NODES + 255) / 256), dim3(256), 0, stream>>>(out, gamma, beta, avg);
}
